// TECAN_50087908606430
// MI455X (gfx1250) — matrix operand order repaired and checked by reading, not compiled, not run on hardware
//
#include <hip/hip_runtime.h>

// ---------------- problem constants ----------------
#define N_      180
#define NP      192          // N padded to 12 WMMA tiles
#define KTOT    540          // 3*N
#define KPAD    544          // 17 K-tiles of 32
#define KS      552          // LDS row stride (halfs) for A and Wcat (16B-aligned rows)
#define MB      16           // batches per workgroup (one WMMA M tile)
#define TB      1024         // timesteps
#define BATCH   256
#define DT_     0.02f
#define K_      0.0005f
#define RHO_W_  2.0f         // 1/RHO_DEG
#define W_VS_   0.2f
#define NTHREADS 384
#define NPAIR   (N_ / 2)     // 90 neuron pairs per batch for E2

typedef __attribute__((ext_vector_type(16))) _Float16 v16h;
typedef __attribute__((ext_vector_type(8)))  _Float16 v8h;
typedef __attribute__((ext_vector_type(2)))  _Float16 v2h;
typedef __attribute__((ext_vector_type(8)))  float    v8f;
typedef __attribute__((ext_vector_type(4)))  float    v4f;

union U16 { v16h v; v8h h[2]; };

// ---------------- LDS layout (bytes) ----------------
// State stored TRANSPOSED: Us[i][b] (i = neuron 0..191, b = batch 0..15) so the
// post-WMMA update reads/writes contiguous 32B runs per lane (ds_load_b128).
constexpr size_t OFF_W    = 0;                                // Wcat f16 [NP][KS]
constexpr size_t SZ_W     = (size_t)NP * KS * 2;              // 211968
constexpr size_t OFF_AF   = OFF_W + SZ_W;                     // Rcat f16 [MB][KS]
constexpr size_t SZ_AF    = (size_t)MB * KS * 2;              // 17664
constexpr size_t OFF_US   = OFF_AF + SZ_AF;                   // f32 [NP][MB]
constexpr size_t OFF_UP   = OFF_US + (size_t)NP * MB * 4;
constexpr size_t OFF_UM   = OFF_UP + (size_t)NP * MB * 4;
constexpr size_t OFF_RO   = OFF_UM + (size_t)NP * MB * 4;     // readout f32 [NP]
constexpr size_t OFF_PS   = OFF_RO + (size_t)NP * 4;          // f32 [MB*16]
constexpr size_t OFF_PD   = OFF_PS + (size_t)MB * 16 * 4;     // f32 [MB*16]
constexpr size_t OFF_INV  = OFF_PD + (size_t)MB * 16 * 4;     // f32 [MB]  1/(1+K*rho*sum)
constexpr size_t OFF_VSH  = OFF_INV + (size_t)MB * 4;         // f32 [MB]
constexpr size_t SMEM_BYTES = OFF_VSH + (size_t)MB * 4;       // ~268 KB < 320 KB WGP LDS

extern "C" __global__ __launch_bounds__(NTHREADS)
void tecan_ring_attractor(const float* __restrict__ vel,       // [B,T,1]
                          const float* __restrict__ B_v,       // [1]
                          const float* __restrict__ readout_W, // [1,N]
                          const float* __restrict__ W_r,       // [N,N]
                          const float* __restrict__ W_plus,    // [N,N]
                          const float* __restrict__ W_minus,   // [N,N]
                          const float* __restrict__ h_init,    // [3N]
                          float* __restrict__ out)             // [B,T,1]
{
    extern __shared__ char smem[];
    _Float16* Wlds = (_Float16*)(smem + OFF_W);
    _Float16* Af   = (_Float16*)(smem + OFF_AF);
    float* Us   = (float*)(smem + OFF_US);   // [i][b]
    float* Up   = (float*)(smem + OFF_UP);   // [i][b]
    float* Um   = (float*)(smem + OFF_UM);   // [i][b]
    float* ro   = (float*)(smem + OFF_RO);
    float* psum = (float*)(smem + OFF_PS);
    float* pdot = (float*)(smem + OFF_PD);
    float* invs = (float*)(smem + OFF_INV);
    float* vsh  = (float*)(smem + OFF_VSH);

    const int tid = threadIdx.x;
    const int b0  = blockIdx.x * MB;
    const float Bv = B_v[0];

    // ---- one-time init: Wcat (f16, stacked [W_r | W_plus | W_minus] along K) ----
    for (int idx = tid; idx < NP * KS; idx += NTHREADS) {
        const int row = idx / KS, k = idx % KS;
        float w = 0.f;
        if (row < N_ && k < KTOT) {
            if (k < N_)          w = W_r    [row * N_ + k];
            else if (k < 2*N_)   w = W_plus [row * N_ + (k - N_)];
            else                 w = W_minus[row * N_ + (k - 2*N_)];
        }
        Wlds[idx] = (_Float16)w;
    }
    for (int idx = tid; idx < MB * KS; idx += NTHREADS) Af[idx] = (_Float16)0.f; // pads stay 0
    for (int idx = tid; idx < NP * MB; idx += NTHREADS) {
        const int i = idx / MB;                 // transposed: [i][b]
        Us[idx] = (i < N_) ? h_init[i]          : 0.f;
        Up[idx] = (i < N_) ? h_init[N_ + i]     : 0.f;
        Um[idx] = (i < N_) ? h_init[2*N_ + i]   : 0.f;
    }
    if (tid < NP) ro[tid] = (tid < N_) ? readout_W[tid] : 0.f;
    __syncthreads();

    const int g    = tid >> 4;    // reduction group (batch) 0..23, active if < MB
    const int j    = tid & 15;    // lane within group
    const int wave = tid >> 5;    // 0..11: column tile owner
    const int lane = tid & 31;
    const int lm   = lane & 15;
    const int half = lane >> 4;
    const int n0   = wave * 16;

    // ---- pre-loop: seed invs[] (t=0 normalization) and vsh[] (t=0 velocity) ----
    if (g < MB) {
        float p = 0.f;
        for (int i = j; i < N_; i += 16) {
            const float u  = Us[i * MB + g];
            const float up = fmaxf(u, 0.f);
            p += up * up;
        }
        psum[g*16 + j] = p;
    }
    if (tid < MB) vsh[tid] = vel[(size_t)(b0 + tid) * TB] * Bv;
    __syncthreads();
    if (g < MB && j == 0) {
        float s = 0.f;
        for (int k = 0; k < 16; ++k) s += psum[g*16 + k];
        invs[g] = 1.f / (1.f + K_ * RHO_W_ * s);
    }
    __syncthreads();

    for (int t = 0; t < TB; ++t) {
        // -- E2: rates -> f16 A staging [r_s | r_p | r_m]; all 384 threads,
        //    packed 2xf16 b32 stores.  Work item = (batch b, neuron pair p) --
        for (int idx = tid; idx < MB * NPAIR; idx += NTHREADS) {
            const int b = idx / NPAIR;
            const int i = (idx - b * NPAIR) * 2;
            const float inv = invs[b];
            const float v   = vsh[b];
            const float cp  = 10.f + v, cm = 10.f - v;
            const float usa = Us[i * MB + b],      usb = Us[(i+1) * MB + b];
            const float upa = Up[i * MB + b],      upb = Up[(i+1) * MB + b];
            const float uma = Um[i * MB + b],      umb = Um[(i+1) * MB + b];
            const float pa = fmaxf(usa, 0.f),      pb = fmaxf(usb, 0.f);
            v2h rs = { (_Float16)(pa * pa * inv),      (_Float16)(pb * pb * inv) };
            v2h rp = { (_Float16)fmaxf(cp * upa, 0.f), (_Float16)fmaxf(cp * upb, 0.f) };
            v2h rm = { (_Float16)fmaxf(cm * uma, 0.f), (_Float16)fmaxf(cm * umb, 0.f) };
            *(v2h*)(Af + b*KS + i)        = rs;
            *(v2h*)(Af + b*KS + N_  + i)  = rp;
            *(v2h*)(Af + b*KS + 2*N_+ i)  = rm;
        }
        __syncthreads();

        // -- G: rec[16x16] tile = Rcat @ Wcat^T (17 x v_wmma_f32_16x16x32_f16) --
        v8f acc = {};
        #pragma unroll
        for (int kt = 0; kt < KPAD / 32; ++kt) {
            U16 ua, ub;
            const int oa = lm * KS + kt * 32 + 8 * half;       // A 16x32 f16 layout
            ua.h[0] = *(const v8h*)(Af + oa);
            ua.h[1] = *(const v8h*)(Af + oa + 16);
            const int ob = (n0 + lm) * KS + kt * 32 + 8 * half;  // B 32x16 f16 layout
            ub.h[0] = *(const v8h*)(Wlds + ob);
            ub.h[1] = *(const v8h*)(Wlds + ob + 16);
            acc = __builtin_amdgcn_wmma_f32_16x16x32_f16(
                false, ua.v, false, ub.v, (short)0, acc, false, false);
        }

        // -- U: Euler update straight from accumulator.
        //    Lane owns neuron i = n0+lm, batches 8*half .. 8*half+7 (contiguous) --
        {
            const int i    = n0 + lm;
            const int base = i * MB + 8 * half;
            v4f us0 = *(v4f*)(Us + base), us1 = *(v4f*)(Us + base + 4);
            v4f up0 = *(v4f*)(Up + base), up1 = *(v4f*)(Up + base + 4);
            v4f um0 = *(v4f*)(Um + base), um1 = *(v4f*)(Um + base + 4);
            v4f iv0 = *(v4f*)(invs + 8 * half), iv1 = *(v4f*)(invs + 8 * half + 4);
            #pragma unroll
            for (int r = 0; r < 8; ++r) {
                const float rec = acc[r];
                const float us  = (r < 4) ? us0[r] : us1[r-4];
                const float up  = (r < 4) ? up0[r] : up1[r-4];
                const float um  = (r < 4) ? um0[r] : um1[r-4];
                const float inv = (r < 4) ? iv0[r] : iv1[r-4];
                const float upos = fmaxf(us, 0.f);
                const float rs = upos * upos * inv;
                const float usn = us + DT_ * (-us + RHO_W_ * rec);
                const float upn = up + DT_ * (-up + W_VS_ * rs);
                const float umn = um + DT_ * (-um + W_VS_ * rs);
                if (r < 4) { us0[r] = usn; up0[r] = upn; um0[r] = umn; }
                else       { us1[r-4] = usn; up1[r-4] = upn; um1[r-4] = umn; }
            }
            *(v4f*)(Us + base) = us0;  *(v4f*)(Us + base + 4) = us1;
            *(v4f*)(Up + base) = up0;  *(v4f*)(Up + base + 4) = up1;
            *(v4f*)(Um + base) = um0;  *(v4f*)(Um + base + 4) = um1;
        }
        __syncthreads();

        // -- O: readout partials on h_new; doubles as next step's normalization sum.
        //    Also prefetch next step's velocity (vsh is dead after E2). --
        if (tid < MB && t + 1 < TB)
            vsh[tid] = vel[(size_t)(b0 + tid) * TB + (t + 1)] * Bv;
        if (g < MB) {
            float ps = 0.f, pd = 0.f;
            for (int i = j; i < N_; i += 16) {
                const float u    = Us[i * MB + g];
                const float upos = fmaxf(u, 0.f);
                const float s2   = upos * upos;
                ps += s2;
                pd += s2 * ro[i];
            }
            psum[g*16 + j] = ps;
            pdot[g*16 + j] = pd;
        }
        __syncthreads();
        if (g < MB && j == 0) {
            float S = 0.f, D = 0.f;
            for (int k = 0; k < 16; ++k) { S += psum[g*16 + k]; D += pdot[g*16 + k]; }
            const float invn = 1.f / (1.f + K_ * RHO_W_ * S);  // next step's normalizer
            invs[g] = invn;
            out[(size_t)(b0 + g) * TB + t] = D * invn;         // readout of this step
        }
        __syncthreads();
    }
}

extern "C" void kernel_launch(void* const* d_in, const int* in_sizes, int n_in,
                              void* d_out, int out_size, void* d_ws, size_t ws_size,
                              hipStream_t stream) {
    (void)in_sizes; (void)n_in; (void)d_ws; (void)ws_size; (void)out_size;
    const float* vel       = (const float*)d_in[0];
    const float* B_v       = (const float*)d_in[1];
    const float* readout_W = (const float*)d_in[2];
    const float* W_r       = (const float*)d_in[3];
    const float* W_plus    = (const float*)d_in[4];
    const float* W_minus   = (const float*)d_in[5];
    const float* h_init    = (const float*)d_in[6];
    float* out = (float*)d_out;

    dim3 grid(BATCH / MB);     // 16 workgroups, one per 16-batch tile
    dim3 block(NTHREADS);      // 12 wave32 waves
    hipLaunchKernelGGL(tecan_ring_attractor, grid, block, SMEM_BYTES, stream,
                       vel, B_v, readout_W, W_r, W_plus, W_minus, h_init, out);
}
